// MMN_29368986370493
// MI455X (gfx1250) — matrix operand order repaired and checked by reading, not compiled, not run on hardware
//
#include <hip/hip_runtime.h>
#include <hip/hip_bf16.h>
#include <math.h>

// ---------------------------------------------------------------------------
// CDNA5 (gfx1250) wave32 WMMA types
// ---------------------------------------------------------------------------
typedef __bf16 v16bf __attribute__((ext_vector_type(16)));
typedef __bf16 v8bf  __attribute__((ext_vector_type(8)));
typedef __bf16 v2bf  __attribute__((ext_vector_type(2)));
typedef float  v8f   __attribute__((ext_vector_type(8)));

#define CAT16(lo, hi) \
    __builtin_shufflevector(lo, hi, 0, 1, 2, 3, 4, 5, 6, 7, 8, 9, 10, 11, 12, 13, 14, 15)

#define TEMP_SCALE 20.0f
#define EPS_COS    1e-8f
#define EPS_NORM   1e-5f

#define HW   4096   // 64*64
#define NB   2      // batch
#define CIN  1024
#define RD   256
#define CO   128

static __device__ __forceinline__ float wave16_max(float v) {
    v = fmaxf(v, __shfl_xor(v, 1, 32));
    v = fmaxf(v, __shfl_xor(v, 2, 32));
    v = fmaxf(v, __shfl_xor(v, 4, 32));
    v = fmaxf(v, __shfl_xor(v, 8, 32));
    return v;
}
static __device__ __forceinline__ float wave16_sum(float v) {
    v += __shfl_xor(v, 1, 32);
    v += __shfl_xor(v, 2, 32);
    v += __shfl_xor(v, 4, 32);
    v += __shfl_xor(v, 8, 32);
    return v;
}
static __device__ __forceinline__ v2bf pack_bf16(float a, float b) {
    v2bf p;
    p[0] = (__bf16)a;
    p[1] = (__bf16)b;
    return p;
}

// ---------------------------------------------------------------------------
// Generic bf16-WMMA GEMM:  C[z] = act(A * B[z] + bias + residual[z])
//   A : M x K row-major (weights, shared over z); B : K x N, stride strideB
//   C : M x N row-major, stride strideC (residual same layout)
// Block tile 64(M) x 128(N), K-step 32; 8 waves, each computes a 2x2 quad of
// 16x16 WMMA tiles (fragments reused twice -> 4 wmma per 8 ds_load_b128).
// Double-buffered LDS with global->reg prefetch of tile k+1 before the math,
// plus global_prefetch of tile k+2 toward L2.
// LDS is fragment-major (A [m][k], B transposed [n][k]); rows padded to 80B
// so the 16-lane b128 fragment reads are bank-conflict-free; bf16 pairs are
// staged with packed ds_store_b32.
// ---------------------------------------------------------------------------
__global__ __launch_bounds__(256)
void gemm_bf16_kernel(const float* __restrict__ A, const float* __restrict__ Bm,
                      const float* __restrict__ bias, const float* __restrict__ res,
                      float* __restrict__ C, int M, int N, int K,
                      long long strideB, long long strideC, int relu) {
    __shared__ __attribute__((aligned(16))) __bf16 a_lds[2][64][40];
    __shared__ __attribute__((aligned(16))) __bf16 b_ldsT[2][128][40];

    const int tid  = threadIdx.x;
    const int wv   = tid >> 5;
    const int lane = tid & 31;
    const int half = lane >> 4;
    const int lx   = lane & 15;

    const int bM = blockIdx.y * 64;
    const int bN = blockIdx.x * 128;
    const float* Bp = Bm + (size_t)blockIdx.z * strideB;

    const int wm = wv >> 2;     // 0..1 -> 32-row slab
    const int wn = wv & 3;      // 0..3 -> 32-col slab
    const int nK = K >> 5;

    float2 ra[4];
    float  rb0[8], rb1[8];
    auto stage_load = [&](int kt) {
        const int k0 = kt * 32;
#pragma unroll
        for (int i = 0; i < 4; ++i) {          // A: 64 x 32 as 1024 pairs
            int idx = i * 256 + tid, m = idx >> 4, kp = idx & 15;
            ra[i] = *(const float2*)&A[(size_t)(bM + m) * K + k0 + 2 * kp];
        }
#pragma unroll
        for (int i = 0; i < 8; ++i) {          // B: 32 x 128 as 2048 pairs
            int idx = i * 256 + tid, n = idx & 127, kp = idx >> 7;
            rb0[i] = Bp[(size_t)(k0 + 2 * kp) * N + bN + n];
            rb1[i] = Bp[(size_t)(k0 + 2 * kp + 1) * N + bN + n];
        }
    };
    auto stage_store = [&](int buf) {
#pragma unroll
        for (int i = 0; i < 4; ++i) {
            int idx = i * 256 + tid, m = idx >> 4, kp = idx & 15;
            *(v2bf*)&a_lds[buf][m][2 * kp] = pack_bf16(ra[i].x, ra[i].y);
        }
#pragma unroll
        for (int i = 0; i < 8; ++i) {
            int idx = i * 256 + tid, n = idx & 127, kp = idx >> 7;
            *(v2bf*)&b_ldsT[buf][n][2 * kp] = pack_bf16(rb0[i], rb1[i]);
        }
    };

    v8f acc[2][2];
#pragma unroll
    for (int mi = 0; mi < 2; ++mi)
#pragma unroll
        for (int ni = 0; ni < 2; ++ni)
#pragma unroll
            for (int r = 0; r < 8; ++r) acc[mi][ni][r] = 0.0f;

    stage_load(0);
    stage_store(0);
    __syncthreads();

    for (int kt = 0; kt < nK; ++kt) {
        const int cur = kt & 1;
        if (kt + 1 < nK) stage_load(kt + 1);   // VMEM issued before matrix math
        if (kt + 2 < nK) {
            // pull tile k+2 toward L2 (global_prefetch_b8; no counter traffic)
            const int k0p = (kt + 2) * 32;
            __builtin_prefetch(&Bp[(size_t)(k0p + (tid >> 3)) * N + bN + ((tid & 7) << 4)], 0, 1);
            __builtin_prefetch(&A[(size_t)(bM + (tid & 63)) * K + k0p], 0, 1);
        }

        v16bf afr[2], bfr[2];
        const int kb = half * 8;
#pragma unroll
        for (int t = 0; t < 2; ++t) {
            const __bf16* arow = &a_lds[cur][wm * 32 + t * 16 + lx][0];
            v8bf a0 = *(const v8bf*)(arow + kb);
            v8bf a1 = *(const v8bf*)(arow + 16 + kb);
            afr[t] = CAT16(a0, a1);
        }
#pragma unroll
        for (int t = 0; t < 2; ++t) {
            const __bf16* brow = &b_ldsT[cur][wn * 32 + t * 16 + lx][0];
            v8bf b0 = *(const v8bf*)(brow + half * 8);
            v8bf b1 = *(const v8bf*)(brow + 16 + half * 8);
            bfr[t] = CAT16(b0, b1);
        }
#pragma unroll
        for (int mi = 0; mi < 2; ++mi)
#pragma unroll
            for (int ni = 0; ni < 2; ++ni)
                acc[mi][ni] = __builtin_amdgcn_wmma_f32_16x16x32_bf16(
                    false, afr[mi], false, bfr[ni], (short)0, acc[mi][ni], false, false);

        if (kt + 1 < nK) stage_store((kt + 1) & 1);
        __syncthreads();
    }

    float* Cp = C + (size_t)blockIdx.z * strideC;
    const float* Rp = res ? (res + (size_t)blockIdx.z * strideC) : nullptr;
#pragma unroll
    for (int mi = 0; mi < 2; ++mi) {
        const int row0 = bM + wm * 32 + mi * 16 + half * 8;
#pragma unroll
        for (int ni = 0; ni < 2; ++ni) {
            const int col = bN + wn * 32 + ni * 16 + lx;
#pragma unroll
            for (int r = 0; r < 8; ++r) {
                float v = acc[mi][ni][r];
                if (bias) v += bias[row0 + r];
                if (Rp)   v += Rp[(size_t)(row0 + r) * N + col];
                if (relu) v = fmaxf(v, 0.0f);
                Cp[(size_t)(row0 + r) * N + col] = v;
            }
        }
    }
}

// ---------------------------------------------------------------------------
// WeightAverage 3x3 cosine-sim window attention (replicate-padded windows).
// Thread-per-pixel, channel loops -> all global accesses coalesced across
// threads.  theta is read once per pixel (all 9 neighbor dots per channel).
// ---------------------------------------------------------------------------
__global__ __launch_bounds__(256)
void wa_window_attn_kernel(const float* __restrict__ th, const float* __restrict__ ph,
                           const float* __restrict__ gg, float* __restrict__ wav) {
    const int p  = blockIdx.x * 256 + threadIdx.x;  // [0, NB*HW)
    const int b  = p >> 12;
    const int hw = p & 4095;
    const int h  = hw >> 6;
    const int w  = hw & 63;
    const size_t base = (size_t)b * CO * HW;

    int pn[9];
#pragma unroll
    for (int k = 0; k < 9; ++k) {
        const int hn = min(max(h + k / 3 - 1, 0), 63);
        const int wn = min(max(w + k % 3 - 1, 0), 63);
        pn[k] = hn * 64 + wn;
    }

    float d[9], n2[9], nt2 = 0.0f;
#pragma unroll
    for (int k = 0; k < 9; ++k) { d[k] = 0.0f; n2[k] = 0.0f; }
#pragma unroll 4
    for (int c = 0; c < CO; ++c) {
        const float tv = th[base + (size_t)c * HW + hw];
        nt2 += tv * tv;
#pragma unroll
        for (int k = 0; k < 9; ++k) {
            const float pv = ph[base + (size_t)c * HW + pn[k]];
            d[k]  += tv * pv;
            n2[k] += pv * pv;
        }
    }
    const float ntheta = fmaxf(sqrtf(nt2), EPS_COS);

    float cs[9];
#pragma unroll
    for (int k = 0; k < 9; ++k)
        cs[k] = d[k] / (fmaxf(sqrtf(n2[k]), EPS_COS) * ntheta);

    float m = cs[0];
#pragma unroll
    for (int k = 1; k < 9; ++k) m = fmaxf(m, cs[k]);
    float att[9], se = 0.0f;
#pragma unroll
    for (int k = 0; k < 9; ++k) { att[k] = __expf(cs[k] - m); se += att[k]; }
    const float inv = 1.0f / se;
#pragma unroll
    for (int k = 0; k < 9; ++k) att[k] *= inv;

#pragma unroll 4
    for (int c = 0; c < CO; ++c) {
        float acc = 0.0f;
#pragma unroll
        for (int k = 0; k < 9; ++k)
            acc += att[k] * gg[base + (size_t)c * HW + pn[k]];
        wav[base + (size_t)c * HW + hw] = acc;
    }
}

// ---------------------------------------------------------------------------
// In-place per-pixel L2 channel normalization: x /= (||x||_2 + 1e-5).
// Thread-per-pixel; both passes fully coalesced (2nd pass hits L2).
// ---------------------------------------------------------------------------
__global__ __launch_bounds__(256)
void l2norm_kernel(float* __restrict__ x) {
    const int p  = blockIdx.x * 256 + threadIdx.x;  // [0, NB*HW)
    const int b  = p >> 12;
    const int hw = p & 4095;
    float* xp = x + (size_t)b * RD * HW + hw;

    float s = 0.0f;
#pragma unroll 8
    for (int c = 0; c < RD; ++c) {
        const float v = xp[(size_t)c * HW];
        s += v * v;
    }
    const float inv = 1.0f / (sqrtf(s) + EPS_NORM);
#pragma unroll 8
    for (int c = 0; c < RD; ++c)
        xp[(size_t)c * HW] *= inv;
}

// ---------------------------------------------------------------------------
// Fused correlation -> softmax(*20) -> f_s aggregation (flash-attention).
// Block = 128 queries (8 waves x 16).  Wave keeps Q fragments (8 x v16bf) and
// the 256-channel output accumulators (16 x v8f) in registers; streams support
// in 32-wide tiles: score WMMA (K=256), online softmax, value WMMA (K=32).
// LDS tiles are fragment-major (2x ds_load_b128 per operand) and staged as
// packed bf16 pairs (ds_store_b32); next tile is prefetched toward L2.
// ---------------------------------------------------------------------------
__global__ __launch_bounds__(256)
void flash_corr_attn_kernel(const float* __restrict__ fqn, const float* __restrict__ fsn,
                            const float* __restrict__ fsv, float* __restrict__ att) {
    const int b    = blockIdx.z;
    const int tid  = threadIdx.x;
    const int wv   = tid >> 5;
    const int lane = tid & 31;
    const int half = lane >> 4;
    const int lx   = lane & 15;

    const float* Q = fqn + (size_t)b * RD * HW;
    const float* S = fsn + (size_t)b * RD * HW;
    const float* V = fsv + (size_t)b * RD * HW;
    float* O       = att + (size_t)b * HW * RD;

    const int qb = blockIdx.x * 128 + wv * 16;

    __shared__ __attribute__((aligned(16))) __bf16 s_t[32][264];     // fsn^T tile: s x c
    __shared__ __attribute__((aligned(16))) __bf16 v_t2[RD][40];     // f_s tile: c x s
    __shared__ __attribute__((aligned(16))) __bf16 p_lds[8][16][40]; // per-wave P

    // Preload Q^T fragments for all 8 K-steps (K = 256).
    v16bf aq[8];
#pragma unroll
    for (int kc = 0; kc < 8; ++kc) {
        const int q  = qb + lx;
        const int kb = kc * 32 + half * 8;
#pragma unroll
        for (int i = 0; i < 8; ++i) {
            aq[kc][i]     = (__bf16)Q[(size_t)(kb + i) * HW + q];
            aq[kc][8 + i] = (__bf16)Q[(size_t)(kb + 16 + i) * HW + q];
        }
    }

    v8f o[16];
#pragma unroll
    for (int cc = 0; cc < 16; ++cc)
#pragma unroll
        for (int r = 0; r < 8; ++r) o[cc][r] = 0.0f;
    float mrun[8], lrun[8];
#pragma unroll
    for (int r = 0; r < 8; ++r) { mrun[r] = -3.0e38f; lrun[r] = 0.0f; }

    for (int s0 = 0; s0 < HW; s0 += 32) {
        __syncthreads();
        // Stage support tiles (fp32 -> packed bf16 pairs -> ds_store_b32).
        {
            const int cA = tid >> 4;            // v_t2: pair along s
            const int jA = (tid & 15) * 2;
            const int jB = tid & 31;            // s_t: pair along c
            const int cB = (tid >> 5) * 2;
#pragma unroll
            for (int rr = 0; rr < 16; ++rr) {
                const int c = rr * 16 + cA;
                const float2 vv = *(const float2*)&V[(size_t)c * HW + s0 + jA];
                *(v2bf*)&v_t2[c][jA] = pack_bf16(vv.x, vv.y);

                const int c0 = rr * 16 + cB;
                const float f0 = S[(size_t)c0 * HW + s0 + jB];
                const float f1 = S[(size_t)(c0 + 1) * HW + s0 + jB];
                *(v2bf*)&s_t[jB][c0] = pack_bf16(f0, f1);
            }
        }
        if (s0 + 32 < HW) {
            // pull next support tile toward L2 (one 128B row per thread/array)
            __builtin_prefetch(&S[(size_t)tid * HW + s0 + 32], 0, 1);
            __builtin_prefetch(&V[(size_t)tid * HW + s0 + 32], 0, 1);
        }
        __syncthreads();

        // -------- GEMM1: scores 16q x 32s, K = 256 --------
        v8f sc0, sc1;
#pragma unroll
        for (int r = 0; r < 8; ++r) { sc0[r] = 0.0f; sc1[r] = 0.0f; }
        const __bf16* sr0 = &s_t[lx][0];
        const __bf16* sr1 = &s_t[16 + lx][0];
#pragma unroll
        for (int kc = 0; kc < 8; ++kc) {
            const int k0 = kc * 32 + half * 8;
            v8bf x0 = *(const v8bf*)(sr0 + k0);
            v8bf x1 = *(const v8bf*)(sr0 + k0 + 16);
            v8bf y0 = *(const v8bf*)(sr1 + k0);
            v8bf y1 = *(const v8bf*)(sr1 + k0 + 16);
            v16bf b0 = CAT16(x0, x1);
            v16bf b1 = CAT16(y0, y1);
            sc0 = __builtin_amdgcn_wmma_f32_16x16x32_bf16(
                false, aq[kc], false, b0, (short)0, sc0, false, false);
            sc1 = __builtin_amdgcn_wmma_f32_16x16x32_bf16(
                false, aq[kc], false, b1, (short)0, sc1, false, false);
        }

        // -------- online softmax over the 32-wide s tile --------
        float p0[8], p1[8], cf[8];
#pragma unroll
        for (int r = 0; r < 8; ++r) {
            const float x0 = sc0[r] * TEMP_SCALE;
            const float x1 = sc1[r] * TEMP_SCALE;
            float rm = wave16_max(fmaxf(x0, x1));
            const float mo = mrun[r];
            const float mn = fmaxf(mo, rm);
            const float c_ = __expf(mo - mn);
            p0[r] = __expf(x0 - mn);
            p1[r] = __expf(x1 - mn);
            const float rs = wave16_sum(p0[r] + p1[r]);
            lrun[r] = lrun[r] * c_ + rs;
            mrun[r] = mn;
            cf[r]   = c_;
        }
#pragma unroll
        for (int cc = 0; cc < 16; ++cc)
#pragma unroll
            for (int r = 0; r < 8; ++r) o[cc][r] *= cf[r];

        // -------- P (C-layout) -> A-fragment layout via LDS --------
#pragma unroll
        for (int r = 0; r < 8; ++r) {
            p_lds[wv][half * 8 + r][lx]      = (__bf16)p0[r];
            p_lds[wv][half * 8 + r][16 + lx] = (__bf16)p1[r];
        }
        __syncthreads();
        const __bf16* pr = &p_lds[wv][lx][0];
        const int kb2 = half * 8;
        v8bf q0 = *(const v8bf*)(pr + kb2);
        v8bf q1 = *(const v8bf*)(pr + 16 + kb2);
        v16bf ap = CAT16(q0, q1);

        // -------- GEMM2: O[16q x 256c] += P(16x32) * V^T(32xc) --------
#pragma unroll
        for (int cc = 0; cc < 16; ++cc) {
            const __bf16* vr = &v_t2[cc * 16 + lx][0];
            v8bf z0 = *(const v8bf*)(vr + half * 8);
            v8bf z1 = *(const v8bf*)(vr + 16 + half * 8);
            v16bf bv = CAT16(z0, z1);
            o[cc] = __builtin_amdgcn_wmma_f32_16x16x32_bf16(
                false, ap, false, bv, (short)0, o[cc], false, false);
        }
    }

    // Epilogue: divide by softmax denominators, store (q-major, c contiguous).
#pragma unroll
    for (int r = 0; r < 8; ++r) {
        const float inv = 1.0f / lrun[r];
        const int q = qb + half * 8 + r;
#pragma unroll
        for (int cc = 0; cc < 16; ++cc)
            O[(size_t)q * RD + cc * 16 + lx] = o[cc][r] * inv;
    }
}

// ---------------------------------------------------------------------------
// Batch mean + transpose (q-major att -> channel-major out) via 64x64 LDS
// tile; coalesced on both the read and write side.
// ---------------------------------------------------------------------------
__global__ __launch_bounds__(256)
void batch_mean_kernel(const float* __restrict__ att, float* __restrict__ out) {
    __shared__ float t_lds[64][65];
    const int tid = threadIdx.x;
    const int q0 = blockIdx.x * 64;
    const int c0 = blockIdx.y * 64;
#pragma unroll
    for (int i = 0; i < 16; ++i) {
        const int idx = i * 256 + tid;
        const int qq = idx >> 6, cc = idx & 63;
        const float a0 = att[(size_t)(q0 + qq) * RD + c0 + cc];
        const float a1 = att[(size_t)HW * RD + (size_t)(q0 + qq) * RD + c0 + cc];
        t_lds[qq][cc] = 0.5f * (a0 + a1);
    }
    __syncthreads();
#pragma unroll
    for (int i = 0; i < 16; ++i) {
        const int idx = i * 256 + tid;
        const int cc = idx >> 6, qq = idx & 63;
        out[(size_t)(c0 + cc) * HW + q0 + qq] = t_lds[qq][cc];
    }
}

// ---------------------------------------------------------------------------
// Host-side orchestration
// ---------------------------------------------------------------------------
static void gemm_launch(const float* A, const float* B, const float* bias,
                        const float* res, float* C, int M, int N, int K,
                        long long sB, long long sC, int relu, int batch,
                        hipStream_t stream) {
    dim3 grid(N / 128, M / 64, batch), block(256);
    gemm_bf16_kernel<<<grid, block, 0, stream>>>(A, B, bias, res, C, M, N, K, sB, sC, relu);
}

extern "C" void kernel_launch(void* const* d_in, const int* in_sizes, int n_in,
                              void* d_out, int out_size, void* d_ws, size_t ws_size,
                              hipStream_t stream) {
    (void)in_sizes; (void)n_in; (void)out_size; (void)ws_size;
    const float* fq   = (const float*)d_in[0];
    const float* fs   = (const float*)d_in[1];
    const float* f_s  = (const float*)d_in[2];
    const float* rd_w = (const float*)d_in[3];
    const float* wt   = (const float*)d_in[4];
    const float* bt   = (const float*)d_in[5];
    const float* wp   = (const float*)d_in[6];
    const float* bp   = (const float*)d_in[7];
    const float* wg   = (const float*)d_in[8];
    const float* bg   = (const float*)d_in[9];
    const float* wb   = (const float*)d_in[10];
    const float* bb   = (const float*)d_in[11];
    float* out = (float*)d_out;

    float* ws = (float*)d_ws;
    float* fqr  = ws;                     // (2,256,4096)
    float* fsr  = ws + 2097152;           // (2,256,4096)
    float* feaq = ws + 4194304;           // (2,256,4096)
    float* feas = ws + 6291456;           // (2,256,4096)
    float* th   = ws + 8388608;           // (2,128,4096)
    float* ph   = ws + 9437184;           // (2,128,4096)
    float* gg   = ws + 10485760;          // (2,128,4096)
    float* wav  = ws + 11534336;          // (2,128,4096)
    float* attb = ws + 12582912;          // (2,4096,256)

    const long long sIn  = (long long)CIN * HW;   // 1024*4096
    const long long sRD  = (long long)RD * HW;    // 256*4096
    const long long sCO  = (long long)CO * HW;    // 128*4096

    // 1) reduce-dim 1x1 conv + ReLU (shared weights, batched over z)
    gemm_launch(rd_w, fq, nullptr, nullptr, fqr, RD, HW, CIN, sIn, sRD, 1, NB, stream);
    gemm_launch(rd_w, fs, nullptr, nullptr, fsr, RD, HW, CIN, sIn, sRD, 1, NB, stream);

    // 2) WeightAverage on the query branch
    gemm_launch(wt, fqr, bt, nullptr, th, CO, HW, RD, sRD, sCO, 0, NB, stream);
    gemm_launch(wp, fqr, bp, nullptr, ph, CO, HW, RD, sRD, sCO, 0, NB, stream);
    gemm_launch(wg, fqr, bg, nullptr, gg, CO, HW, RD, sRD, sCO, 0, NB, stream);
    wa_window_attn_kernel<<<dim3(NB * HW / 256), dim3(256), 0, stream>>>(th, ph, gg, wav);
    gemm_launch(wb, wav, bb, fqr, feaq, RD, HW, CO, sCO, sRD, 0, NB, stream);

    // 3) WeightAverage on the support branch (reuse th/ph/gg/wav scratch)
    gemm_launch(wt, fsr, bt, nullptr, th, CO, HW, RD, sRD, sCO, 0, NB, stream);
    gemm_launch(wp, fsr, bp, nullptr, ph, CO, HW, RD, sRD, sCO, 0, NB, stream);
    gemm_launch(wg, fsr, bg, nullptr, gg, CO, HW, RD, sRD, sCO, 0, NB, stream);
    wa_window_attn_kernel<<<dim3(NB * HW / 256), dim3(256), 0, stream>>>(th, ph, gg, wav);
    gemm_launch(wb, wav, bb, fsr, feas, RD, HW, CO, sCO, sRD, 0, NB, stream);

    // 4) channel L2 normalization (in place)
    l2norm_kernel<<<dim3(NB * HW / 256), dim3(256), 0, stream>>>(feaq);
    l2norm_kernel<<<dim3(NB * HW / 256), dim3(256), 0, stream>>>(feas);

    // 5) fused corr -> softmax(*20) -> f_s aggregation (never materializes corr)
    flash_corr_attn_kernel<<<dim3(HW / 128, 1, NB), dim3(256), 0, stream>>>(feaq, feas, f_s, attb);

    // 6) batch mean + transpose -> (1, 256, 64, 64)
    batch_mean_kernel<<<dim3(HW / 64, RD / 64), dim3(256), 0, stream>>>(attb, out);
}
